// SlotSetEncoder_77309412084
// MI455X (gfx1250) — compile-verified, matrix operand order checked by reading, not run on hardware
//
#include <hip/hip_runtime.h>
#include <hip/hip_bf16.h>

// ---------------------------------------------------------------------------
// SlotSetEncoder for MI455X (gfx1250): wave32, bf16 WMMA, TDM double-buffered
// X streaming. D=256, H=4, dh=64, K_SLOTS=64; softmax over slot axis factors
// into global accumulators -> single streaming pass over X.
// ---------------------------------------------------------------------------

typedef __attribute__((ext_vector_type(16))) __bf16         v16bf;
typedef __attribute__((ext_vector_type(8)))  __bf16         v8bf;
typedef __attribute__((ext_vector_type(8)))  float          v8f;
typedef __attribute__((ext_vector_type(8)))  unsigned short us8;
typedef __attribute__((ext_vector_type(4)))  unsigned int   v4u;
typedef __attribute__((ext_vector_type(8)))  int            v8i;
typedef __attribute__((ext_vector_type(4)))  int            v4i;

#define DIMF 256
#define NSLOT 64
#define STILE 64
#define TPW 4            // sequence tiles per workgroup (register-resident AV acc)

#ifndef __has_builtin
#define __has_builtin(x) 0
#endif
#if __has_builtin(__builtin_amdgcn_tensor_load_to_lds) && __has_builtin(__builtin_amdgcn_s_wait_tensorcnt)
#define HAS_TDM 1
#else
#define HAS_TDM 0
#endif

// LDS layout (bytes), dynamic shared, total 256 KB
#define OFF_XF0 0              // [64][256] f32 DMA stage, buffer 0 (65536 B)
#define OFF_XF1 65536          // [64][256] f32 DMA stage, buffer 1 (65536 B)
#define OFF_XS  131072         // [64][264] bf16 X tile              (33792 B)
#define OFF_KS  164864         // [64][264] bf16 Kp tile             (33792 B)
#define OFF_VT  198656         // [256][72] bf16 V tile, transposed  (36864 B)
#define OFF_PS  235520         // [64][68]  f32 scores/probs         (17408 B)
#define OFF_PB  252928         // [64][72]  bf16 probs               ( 9216 B)
#define LDS_BYTES 262144

// float -> bf16 (round to nearest even), bit-level
__device__ __forceinline__ unsigned short f2bf(float f) {
  unsigned int u = __float_as_uint(f);
  u = (u + 0x7FFFu + ((u >> 16) & 1u)) >> 16;
  return (unsigned short)u;
}

// 16x32 bf16 WMMA operand fragment from a row-major matrix (A pattern; also B
// when stored n-major with k contiguous per row).
__device__ __forceinline__ v16bf frag16x32(const unsigned short* base, int ld,
                                           int row0, int k0, int lane) {
  const int half = lane >> 4;
  const int r    = lane & 15;
  const unsigned short* p = base + (row0 + r) * ld + k0 + half * 8;
  v8bf lo = *(const v8bf*)(p);
  v8bf hi = *(const v8bf*)(p + 16);
  v16bf f;
#pragma unroll
  for (int i = 0; i < 8; i++) { f[i] = lo[i]; f[i + 8] = hi[i]; }
  return f;
}

__device__ __forceinline__ v8f wmma_bf16(v16bf a, v16bf b, v8f c) {
  return __builtin_amdgcn_wmma_f32_16x16x32_bf16(false, a, false, b,
                                                 (short)0, c, false, false);
}

#if HAS_TDM
// Issue a TDM load of a [64][256] f32 tile (rows contiguous, stride 256) from
// global `gsrc` into LDS offset `lds_off`. 2D D#: tensor [S][256], tile [64][256].
// amdgpu-toolchain (clang-23) 6-arg form: (v4u, v8i, v4i, v4i, v8i, i32 cpol).
__device__ __forceinline__ void tdm_load_tile(const float* gsrc, unsigned lds_off, int S) {
  unsigned long long ga = (unsigned long long)(const void*)gsrc;
  v4u g0;
  g0[0] = 1u;                                   // count=1, no gather
  g0[1] = lds_off;                              // lds_addr
  g0[2] = (unsigned)(ga & 0xffffffffu);         // global_addr[31:0]
  g0[3] = (unsigned)((ga >> 32) & 0x01ffffffu)  // global_addr[56:32]
        | (2u << 30);                           // type = 2 ("image")
  v8i g1;
  g1[0] = (int)(2u << 16);                      // wg_mask=0, data_size=4B
  g1[1] = (int)(256u << 16);                    // tensor_dim0 = 256 (low 16)
  g1[2] = (int)(((unsigned)S & 0xffffu) << 16); // dim0 hi=0 | tensor_dim1 lo
  g1[3] = (int)(((unsigned)S >> 16) | (256u << 16)); // dim1 hi | tile_dim0=256
  g1[4] = 64;                                   // tile_dim1=64, tile_dim2=0
  g1[5] = 256;                                  // tensor_dim0_stride = 256
  g1[6] = 0;
  g1[7] = 0;
  v4i z4 = {0, 0, 0, 0};
  v8i z8 = {0, 0, 0, 0, 0, 0, 0, 0};
  __builtin_amdgcn_tensor_load_to_lds(g0, g1, z4, z4, z8, 0);
}
#endif

// ---------------------------------------------------------------------------
// Kernel 1: prep. Blocks 0..255 transpose Wk/Wv -> bf16 [d_out][d_in].
// Block 256 computes Q = LN(mu) @ Wq + bq -> bf16 [h][slot][dh].
// ---------------------------------------------------------------------------
__global__ __launch_bounds__(256) void sse_prep(
    const float* __restrict__ mu, const float* __restrict__ lsg,
    const float* __restrict__ lsb, const float* __restrict__ Wq,
    const float* __restrict__ bq, const float* __restrict__ Wk,
    const float* __restrict__ Wv, unsigned short* __restrict__ Qbf,
    unsigned short* __restrict__ WkT, unsigned short* __restrict__ WvT) {
  const int tid = threadIdx.x;
  if (blockIdx.x < 256) {
    const int o = blockIdx.x;
    WkT[o * DIMF + tid] = f2bf(Wk[tid * DIMF + o]);
    WvT[o * DIMF + tid] = f2bf(Wv[tid * DIMF + o]);
    return;
  }
  __shared__ float red[256];
  __shared__ float xn[256];
  for (int slot = 0; slot < NSLOT; slot++) {
    float x = mu[slot * DIMF + tid];
    red[tid] = x; __syncthreads();
    for (int o = 128; o > 0; o >>= 1) { if (tid < o) red[tid] += red[tid + o]; __syncthreads(); }
    float mean = red[0] * (1.f / 256.f); __syncthreads();
    float d = x - mean;
    red[tid] = d * d; __syncthreads();
    for (int o = 128; o > 0; o >>= 1) { if (tid < o) red[tid] += red[tid + o]; __syncthreads(); }
    float var = red[0] * (1.f / 256.f); __syncthreads();
    xn[tid] = d * rsqrtf(var + 1e-5f) * lsg[tid] + lsb[tid];
    __syncthreads();
    float acc = bq[tid];
#pragma unroll 8
    for (int i = 0; i < DIMF; i++) acc = fmaf(xn[i], Wq[i * DIMF + tid], acc);
    const int h = tid >> 6, dh = tid & 63;
    Qbf[h * (NSLOT * 64) + slot * 64 + dh] = f2bf(acc);
    __syncthreads();
  }
}

// ---------------------------------------------------------------------------
// Kernel 2: zero accumulators.
// ---------------------------------------------------------------------------
__global__ void sse_zero(float* __restrict__ Nacc, float* __restrict__ Cacc,
                         int nN, int nC) {
  const int i = blockIdx.x * 256 + threadIdx.x;
  if (i < nN) Nacc[i] = 0.f;
  if (i < nC) Cacc[i] = 0.f;
}

// ---------------------------------------------------------------------------
// Kernel 3: main streaming pass. One WG per (batch, 4x64-row sequence chunk).
// TDM double-buffers X tiles into LDS; AV + C accumulators live in registers
// across the 4 tiles and are flushed with atomics once.
// ---------------------------------------------------------------------------
__global__ __launch_bounds__(256) void sse_main(
    const float* __restrict__ X, const unsigned short* __restrict__ Qbf,
    const unsigned short* __restrict__ WkT, const unsigned short* __restrict__ WvT,
    const float* __restrict__ bk, const float* __restrict__ bv,
    float* __restrict__ Nacc, float* __restrict__ Cacc, int S) {
  extern __shared__ char smem[];
  float*          XF0 = (float*)(smem + OFF_XF0);
  float*          XF1 = (float*)(smem + OFF_XF1);
  unsigned short* XS  = (unsigned short*)(smem + OFF_XS);
  unsigned short* KS  = (unsigned short*)(smem + OFF_KS);
  unsigned short* VT  = (unsigned short*)(smem + OFF_VT);
  float*          PS  = (float*)(smem + OFF_PS);
  unsigned short* PB  = (unsigned short*)(smem + OFF_PB);

  const int tid  = threadIdx.x;
  const int lane = tid & 31;
  const int wave = tid >> 5;
  const int nl   = lane & 15;
  const int mh   = (lane >> 4) * 8;
  const int b    = blockIdx.y;
  const int wgs0 = blockIdx.x * (STILE * TPW);

  const float* Xwg = X + ((size_t)b * S + wgs0) * DIMF;

  // Register-resident accumulators across the TPW tiles.
  v8f   avacc[4][2];
  float creg[4] = {0.f, 0.f, 0.f, 0.f};
#pragma unroll
  for (int h = 0; h < 4; h++) { avacc[h][0] = (v8f){}; avacc[h][1] = (v8f){}; }

#if HAS_TDM
  if (wave == 0) tdm_load_tile(Xwg, OFF_XF0, S);   // prologue DMA for tile 0
#endif

  for (int t0 = 0; t0 < TPW; t0++) {
#if HAS_TDM
    if (wave == 0) {
      if (t0 + 1 < TPW) {  // issue next tile's DMA, then retire current one
        tdm_load_tile(Xwg + (size_t)(t0 + 1) * STILE * DIMF,
                      ((t0 + 1) & 1) ? OFF_XF1 : OFF_XF0, S);
        __builtin_amdgcn_s_wait_tensorcnt(1);
      } else {
        __builtin_amdgcn_s_wait_tensorcnt(0);
      }
    }
    __syncthreads();
    const float* src = (t0 & 1) ? XF1 : XF0;       // LDS f32 stage
#else
    const float* src = Xwg + (size_t)t0 * STILE * DIMF;   // direct global
    if (t0 + 1 < TPW)
      __builtin_prefetch((const char*)(src + (size_t)STILE * DIMF) + tid * 64, 0, 1);
#endif

    // ---- convert f32 tile -> bf16 LDS working copy
#pragma unroll 4
    for (int i = tid; i < STILE * DIMF / 4; i += 256) {
      float4 v = ((const float4*)src)[i];
      const int row = (i * 4) >> 8, col = (i * 4) & 255;
      uint2 u;
      u.x = (unsigned)f2bf(v.x) | ((unsigned)f2bf(v.y) << 16);
      u.y = (unsigned)f2bf(v.z) | ((unsigned)f2bf(v.w) << 16);
      *(uint2*)(XS + row * 264 + col) = u;
    }
    __syncthreads();

    // ---- projections: Kp = X@Wk + bk (normal), V = X@Wv + bv (transposed)
    for (int j = wave; j < 128; j += 8) {
      const int mat = j & 1;
      const int t   = j >> 1;
      const int m0  = (t >> 4) * 16;
      const int n0  = (t & 15) * 16;
      const unsigned short* WT = mat ? WvT : WkT;
      v8f acc = {};
#pragma unroll
      for (int kk = 0; kk < 8; kk++) {
        v16bf a  = frag16x32(XS, 264, m0, kk * 32, lane);
        v16bf bb = frag16x32(WT, DIMF, n0, kk * 32, lane);   // L2-resident
        acc = wmma_bf16(a, bb, acc);
      }
      const float bias = (mat ? bv : bk)[n0 + nl];
      if (!mat) {
#pragma unroll
        for (int i = 0; i < 8; i++)
          KS[(m0 + mh + i) * 264 + n0 + nl] = f2bf(acc[i] + bias);
      } else {
        us8 pk;
#pragma unroll
        for (int i = 0; i < 8; i++) pk[i] = f2bf(acc[i] + bias);
        *(us8*)(VT + (n0 + nl) * 72 + m0 + mh) = pk;
      }
    }
    __syncthreads();

    // ---- per head: scores -> slot-softmax -> accumulate C and P@V in regs
    const float scale = 0.125f;  // 1/sqrt(dh)
    for (int h = 0; h < 4; h++) {
      for (int ti = 0; ti < 2; ti++) {
        const int t  = wave + ti * 8;
        const int m0 = (t >> 2) * 16;  // slot tile
        const int n0 = (t & 3) * 16;   // seq tile
        v8f acc = {};
#pragma unroll
        for (int kk = 0; kk < 2; kk++) {
          v16bf a  = frag16x32(Qbf + h * (NSLOT * 64), 64, m0, kk * 32, lane);
          v16bf bb = frag16x32(KS, 264, n0, h * 64 + kk * 32, lane);
          acc = wmma_bf16(a, bb, acc);
        }
#pragma unroll
        for (int i = 0; i < 8; i++)
          PS[(m0 + mh + i) * 68 + n0 + nl] = acc[i] * scale;
      }
      __syncthreads();

      if (tid < STILE) {  // softmax over the 64 slots for this sequence column
        const int s = tid;
        float mx = -3.0e38f;
        for (int k = 0; k < NSLOT; k++) mx = fmaxf(mx, PS[k * 68 + s]);
        float sum = 0.f;
        for (int k = 0; k < NSLOT; k++) {
          float e = __expf(PS[k * 68 + s] - mx);
          PS[k * 68 + s] = e;
          sum += e;
        }
        const float inv = 1.f / sum;
        for (int k = 0; k < NSLOT; k++) {
          float p = PS[k * 68 + s] * inv;
          PS[k * 68 + s] = p;
          PB[k * 72 + s] = f2bf(p);
        }
      }
      __syncthreads();

      if (tid < NSLOT) {  // C[b][h][k] partial: sum over this tile's columns
        float c = 0.f;
        for (int s = 0; s < STILE; s++) c += PS[tid * 68 + s];
        creg[h] += c;
      }

      for (int ti = 0; ti < 2; ti++) {  // P (64x64) @ V_h (64x64)
        const int t  = wave + ti * 8;
        const int m0 = (t >> 2) * 16;  // slot tile
        const int n0 = (t & 3) * 16;   // dh tile
        v8f acc = avacc[h][ti];
#pragma unroll
        for (int kk = 0; kk < 2; kk++) {
          v16bf a  = frag16x32(PB, 72, m0, kk * 32, lane);
          v16bf bb = frag16x32(VT, 72, h * 64 + n0, kk * 32, lane);
          acc = wmma_bf16(a, bb, acc);
        }
        avacc[h][ti] = acc;
      }
      __syncthreads();
    }
  }

  // ---- epilogue: flush register accumulators with atomics (once per WG)
#pragma unroll
  for (int h = 0; h < 4; h++) {
    for (int ti = 0; ti < 2; ti++) {
      const int t  = wave + ti * 8;
      const int m0 = (t >> 2) * 16;
      const int n0 = (t & 3) * 16;
#pragma unroll
      for (int i = 0; i < 8; i++) {
        const int slot = m0 + mh + i;
        atomicAdd(&Nacc[((size_t)b * NSLOT + slot) * DIMF + h * 64 + n0 + nl],
                  avacc[h][ti][i]);
      }
    }
  }
  if (tid < NSLOT) {
#pragma unroll
    for (int h = 0; h < 4; h++)
      atomicAdd(&Cacc[(b * 4 + h) * NSLOT + tid], creg[h]);
  }
}

// ---------------------------------------------------------------------------
// Kernel 4: finalize. A = N / (C + eps), then LayerNorm over D=256.
// ---------------------------------------------------------------------------
__global__ __launch_bounds__(256) void sse_finalize(
    const float* __restrict__ Nacc, const float* __restrict__ Cacc,
    const float* __restrict__ g, const float* __restrict__ bb,
    float* __restrict__ out) {
  __shared__ float red[256];
  const int tid = threadIdx.x;
  const int bs  = blockIdx.x;
  const int b   = bs >> 6, slot = bs & 63;
  const float n = Nacc[(size_t)bs * DIMF + tid];
  const int h   = tid >> 6;
  const float c = Cacc[(b * 4 + h) * NSLOT + slot];
  const float a = n / (c + 1e-20f);
  red[tid] = a; __syncthreads();
  for (int o = 128; o > 0; o >>= 1) { if (tid < o) red[tid] += red[tid + o]; __syncthreads(); }
  const float mean = red[0] * (1.f / 256.f); __syncthreads();
  const float d = a - mean;
  red[tid] = d * d; __syncthreads();
  for (int o = 128; o > 0; o >>= 1) { if (tid < o) red[tid] += red[tid + o]; __syncthreads(); }
  const float var = red[0] * (1.f / 256.f);
  out[(size_t)bs * DIMF + tid] = d * rsqrtf(var + 1e-5f) * g[tid] + bb[tid];
}

// ---------------------------------------------------------------------------
extern "C" void kernel_launch(void* const* d_in, const int* in_sizes, int n_in,
                              void* d_out, int out_size, void* d_ws, size_t ws_size,
                              hipStream_t stream) {
  const float* X    = (const float*)d_in[0];
  const float* mu   = (const float*)d_in[1];
  const float* lsg  = (const float*)d_in[2];
  const float* lsb  = (const float*)d_in[3];
  const float* Wq   = (const float*)d_in[4];
  const float* bq   = (const float*)d_in[5];
  const float* Wk   = (const float*)d_in[6];
  const float* bk   = (const float*)d_in[7];
  const float* Wv   = (const float*)d_in[8];
  const float* bv   = (const float*)d_in[9];
  const float* lag  = (const float*)d_in[10];
  const float* lab  = (const float*)d_in[11];
  float* out = (float*)d_out;

  const int B = 4;
  const int S = in_sizes[0] / (DIMF * B);  // 32768

  // workspace layout
  char* ws = (char*)d_ws;
  unsigned short* Qbf  = (unsigned short*)(ws);           //  32 KB
  unsigned short* WkT  = (unsigned short*)(ws + 32768);   // 128 KB
  unsigned short* WvT  = (unsigned short*)(ws + 163840);  // 128 KB
  float*          Nacc = (float*)(ws + 294912);           // 256 KB
  float*          Cacc = (float*)(ws + 557056);           //   4 KB

  sse_prep<<<257, 256, 0, stream>>>(mu, lsg, lsb, Wq, bq, Wk, Wv, Qbf, WkT, WvT);

  const int nN = B * NSLOT * DIMF;  // 65536
  const int nC = B * 4 * NSLOT;     // 1024
  sse_zero<<<(nN + 255) / 256, 256, 0, stream>>>(Nacc, Cacc, nN, nC);

  dim3 grid(S / (STILE * TPW), B);  // 128 x 4 = 512 workgroups
  sse_main<<<grid, 256, LDS_BYTES, stream>>>(X, Qbf, WkT, WvT, bk, bv, Nacc, Cacc, S);

  sse_finalize<<<B * NSLOT, 256, 0, stream>>>(Nacc, Cacc, lag, lab, out);
}
